// SimpleAttentionV2_26422638805004
// MI455X (gfx1250) — matrix operand order repaired and checked by reading, not compiled, not run on hardware
//
#include <hip/hip_runtime.h>

// ---------------------------------------------------------------------------
// SimpleAttention for MI455X (gfx1250, wave32, WMMA + async-to-LDS)
//   SEQ=8192, D_IN=768, D_OUT=128
//   Kernel 1: Q/K/V projections via v_wmma_f32_16x16x32_f16 (fp32 -> f16 cast)
//   Kernel 2: flash-attention, cooperative double-buffered K/V staging in LDS
//             via GLOBAL_LOAD_ASYNC_TO_LDS_B128 (ASYNCcnt) when available.
// ---------------------------------------------------------------------------

typedef __attribute__((ext_vector_type(16))) _Float16     v16h;
typedef __attribute__((ext_vector_type(8)))  float        v8f;
typedef __attribute__((ext_vector_type(4)))  unsigned int uvec4;
typedef int v4i __attribute__((vector_size(16)));   // matches builtin param type

#define SEQ   8192
#define DIN   768
#define DOUT  128
#define INV_SQRT_D 0.08838834764831845f   // 1/sqrt(128)
#define LOG2E      1.4426950408889634f

#if defined(__gfx1250__) &&                                            \
    __has_builtin(__builtin_amdgcn_global_load_async_to_lds_b128) &&   \
    __has_builtin(__builtin_amdgcn_s_wait_asynccnt)
#define HAVE_ASYNC 1
#else
#define HAVE_ASYNC 0
#endif

union FragA {              // 16 f16 per lane = 8 VGPRs
    v16h     v;
    _Float16 h[16];
    uvec4    q[2];
};

// 16-byte global -> LDS copy (per lane). Async path uses the gfx1250
// async-to-LDS instruction class (tracked by ASYNCcnt).
__device__ __forceinline__ void cp16(const _Float16* g, _Float16* l)
{
#if HAVE_ASYNC
    v4i* gp = (v4i*)g;     // drop const + reinterpret (generic AS)
    v4i* lp = (v4i*)l;
    __builtin_amdgcn_global_load_async_to_lds_b128(
        (__attribute__((address_space(1))) v4i*)gp,
        (__attribute__((address_space(3))) v4i*)lp,
        /*offset=*/0, /*cpol=*/0);
#else
    *(uvec4*)l = *(const uvec4*)g;
#endif
}

#if HAVE_ASYNC
#define WAIT_ASYNC(n) __builtin_amdgcn_s_wait_asynccnt(n)
#else
#define WAIT_ASYNC(n) ((void)0)
#endif

// ---------------------------------------------------------------------------
// Kernel 1: Y = X @ W^T for W in {Wq, Wk, Wv}.
//   One wave computes one 16x16 output tile; K-loop over DIN in steps of 32.
//   A-frag (16x32, f16): lane m=lane&15; elem e -> k = (e>>3)*16 + hi*8 + (e&7)
//   B-frag (32x16, f16): lane n=lane&15; elem e -> k = (e>>3)*16 + hi*8 + (e&7)
//   Q,K stored row-major f16 [SEQ][DOUT]; V stored transposed f16 [DOUT][SEQ].
// ---------------------------------------------------------------------------
__global__ __launch_bounds__(128)
void qkv_kernel(const float* __restrict__ x,
                const float* __restrict__ wq,
                const float* __restrict__ wk,
                const float* __restrict__ wv,
                _Float16* __restrict__ Qh,
                _Float16* __restrict__ Kh,
                _Float16* __restrict__ Vt)
{
    const int lane = threadIdx.x & 31;
    const int wave = threadIdx.x >> 5;
    const int hi   = lane >> 4;
    const int ln   = lane & 15;

    const int tileM = blockIdx.x * 4 + wave;   // 512 row tiles
    const int mat   = blockIdx.y >> 3;         // 0=Q 1=K 2=V
    const int tileN = blockIdx.y & 7;          // 8 col tiles (DOUT/16)

    const float* w = (mat == 0) ? wq : (mat == 1) ? wk : wv;

    v8f c = {};
    const int rowA = tileM * 16 + ln;
    const int rowB = tileN * 16 + ln;

    for (int kk = 0; kk < DIN; kk += 32) {
        FragA a, b;
        const float* ap0 = x + rowA * DIN + kk + hi * 8;
        const float* ap1 = ap0 + 16;
        #pragma unroll
        for (int e = 0; e < 8; ++e) {
            a.h[e]     = (_Float16)ap0[e];
            a.h[8 + e] = (_Float16)ap1[e];
        }
        const float* bp = w + rowB * DIN + kk + hi * 8;
        #pragma unroll
        for (int e = 0; e < 16; ++e) b.h[e] = (_Float16)bp[(e >> 3) * 16 + (e & 7)];

        c = __builtin_amdgcn_wmma_f32_16x16x32_f16(
                false, a.v, false, b.v, (short)0, c, false, false);
    }

    if (mat == 2) {
        // V transposed: Vt[d][seq]; rows r are consecutive seq -> one b128 store
        union { _Float16 h[8]; uvec4 q; } pk;
        #pragma unroll
        for (int r = 0; r < 8; ++r) pk.h[r] = (_Float16)c[r];
        _Float16* dst = Vt + (tileN * 16 + ln) * SEQ + tileM * 16 + hi * 8;
        *(uvec4*)dst = pk.q;
    } else {
        _Float16* dst = (mat == 0) ? Qh : Kh;
        const float s = (mat == 0) ? INV_SQRT_D : 1.0f;
        #pragma unroll
        for (int r = 0; r < 8; ++r)
            dst[(tileM * 16 + hi * 8 + r) * DOUT + tileN * 16 + ln] =
                (_Float16)(c[r] * s);
    }
}

// ---------------------------------------------------------------------------
// Kernel 2: flash attention. 4 waves/block, each owns 16 query rows; block
// cooperatively stages each 32-key K/V chunk into LDS (double-buffered, async)
// so the chunk is loaded from L2 once per block instead of once per wave.
//   K staging:  kbuf[key][d]  32x128 f16 (8 KB)
//   V staging:  vbuf[d][key] 128x 32 f16 (8 KB)  (from transposed Vt)
//   Per chunk per wave: 8 WMMA for S = Q.K^T, online softmax, P -> per-wave
//   LDS re-layout (D-frag -> A-frag), 8 WMMA for O += P.V
// ---------------------------------------------------------------------------
__global__ __launch_bounds__(128)
void attn_kernel(const _Float16* __restrict__ Qh,
                 const _Float16* __restrict__ Kh,
                 const _Float16* __restrict__ Vt,
                 float* __restrict__ out)
{
    __shared__ _Float16 kbuf[2][32 * DOUT];   // 2 x 8 KB
    __shared__ _Float16 vbuf[2][DOUT * 32];   // 2 x 8 KB
    __shared__ _Float16 pbuf[4][16 * 32];     // 1 KB per wave

    const int tid  = threadIdx.x;
    const int lane = tid & 31;
    const int wave = tid >> 5;
    const int hi   = lane >> 4;
    const int ln   = lane & 15;
    const int qtile = blockIdx.x * 4 + wave;   // 512 query tiles
    _Float16* plds = pbuf[wave];

    // Resident Q A-fragments: 4 K-steps covering d = 0..127
    FragA qf[4];
    {
        const _Float16* qbase = Qh + (qtile * 16 + ln) * DOUT;
        #pragma unroll
        for (int kk = 0; kk < 4; ++kk) {
            qf[kk].q[0] = *(const uvec4*)(qbase + kk * 32 + hi * 8);
            qf[kk].q[1] = *(const uvec4*)(qbase + kk * 32 + 16 + hi * 8);
        }
    }

    // Stage one 32-key chunk: each of 128 threads moves 4x16B of K + 4x16B of V
    auto stage = [&](int buf, int kc) {
        #pragma unroll
        for (int j = 0; j < 4; ++j) {
            const int idx  = tid * 4 + j;          // 0..511
            const int key  = idx >> 4;             // 16 transfers per key row
            const int part = idx & 15;
            cp16(Kh + (kc + key) * DOUT + part * 8,
                 &kbuf[buf][key * DOUT + part * 8]);
        }
        #pragma unroll
        for (int j = 0; j < 4; ++j) {
            const int idx  = tid * 4 + j;          // 0..511
            const int d    = idx >> 2;             // 4 transfers per d row
            const int part = idx & 3;
            cp16(Vt + d * SEQ + kc + part * 8,
                 &vbuf[buf][d * 32 + part * 8]);
        }
    };

    v8f o[8] = {};
    float mrow[8], lrow[8];
    #pragma unroll
    for (int r = 0; r < 8; ++r) { mrow[r] = -1e30f; lrow[r] = 0.0f; }

    stage(0, 0);                                  // prime buffer 0

    for (int it = 0, kc = 0; kc < SEQ; kc += 32, ++it) {
        const int cur = it & 1;
        if (kc + 32 < SEQ) {                      // prefetch next chunk
            stage(cur ^ 1, kc + 32);
            WAIT_ASYNC(8);                        // this chunk's 8 ops done,
        } else {                                  // prefetch stays in flight
            WAIT_ASYNC(0);
        }
        __syncthreads();                          // all waves' staging visible

        const _Float16* kb = &kbuf[cur][0];
        const _Float16* vb = &vbuf[cur][0];

        // ---- S = Q . K^T for 32 keys (two 16-key n-tiles), from LDS ----
        v8f s0 = {}, s1 = {};
        #pragma unroll
        for (int kk = 0; kk < 4; ++kk) {
            FragA b0, b1;
            const _Float16* kb0 = kb + ln * DOUT + kk * 32 + hi * 8;
            const _Float16* kb1 = kb0 + 16 * DOUT;
            b0.q[0] = *(const uvec4*)(kb0);
            b0.q[1] = *(const uvec4*)(kb0 + 16);
            b1.q[0] = *(const uvec4*)(kb1);
            b1.q[1] = *(const uvec4*)(kb1 + 16);
            s0 = __builtin_amdgcn_wmma_f32_16x16x32_f16(
                     false, qf[kk].v, false, b0.v, (short)0, s0, false, false);
            s1 = __builtin_amdgcn_wmma_f32_16x16x32_f16(
                     false, qf[kk].v, false, b1.v, (short)0, s1, false, false);
        }

        // ---- online softmax: row = hi*8 + r; 16 cols across 16 lanes ----
        #pragma unroll
        for (int r = 0; r < 8; ++r) {
            float mx = fmaxf(s0[r], s1[r]);
            #pragma unroll
            for (int off = 8; off >= 1; off >>= 1)
                mx = fmaxf(mx, __shfl_xor(mx, off, 16));
            const float mnew = fmaxf(mrow[r], mx);
            const float corr = exp2f((mrow[r] - mnew) * LOG2E);
            const float p0   = exp2f((s0[r] - mnew) * LOG2E);
            const float p1   = exp2f((s1[r] - mnew) * LOG2E);
            float ps = p0 + p1;
            #pragma unroll
            for (int off = 8; off >= 1; off >>= 1)
                ps += __shfl_xor(ps, off, 16);
            lrow[r] = lrow[r] * corr + ps;
            mrow[r] = mnew;
            #pragma unroll
            for (int t = 0; t < 8; ++t) o[t][r] *= corr;
            plds[(hi * 8 + r) * 32 + ln]      = (_Float16)p0;
            plds[(hi * 8 + r) * 32 + 16 + ln] = (_Float16)p1;
        }

        // ---- reload P as A-fragment (per-wave LDS slice) ----
        FragA pf;
        pf.q[0] = *(const uvec4*)(plds + ln * 32 + hi * 8);
        pf.q[1] = *(const uvec4*)(plds + ln * 32 + 16 + hi * 8);

        // ---- O += P . V  from staged LDS ----
        #pragma unroll
        for (int t = 0; t < 8; ++t) {
            FragA vfr;
            const _Float16* vp = vb + (t * 16 + ln) * 32 + hi * 8;
            vfr.q[0] = *(const uvec4*)(vp);
            vfr.q[1] = *(const uvec4*)(vp + 16);
            o[t] = __builtin_amdgcn_wmma_f32_16x16x32_f16(
                       false, pf.v, false, vfr.v, (short)0, o[t], false, false);
        }

        __syncthreads();   // protect buffer from next iteration's prefetch
    }

    // ---- epilogue: divide by row sum, store fp32 ----
    #pragma unroll
    for (int r = 0; r < 8; ++r) {
        const float inv = 1.0f / lrow[r];
        const int row = qtile * 16 + hi * 8 + r;
        #pragma unroll
        for (int t = 0; t < 8; ++t)
            out[row * DOUT + t * 16 + ln] = o[t][r] * inv;
    }
}

// ---------------------------------------------------------------------------
extern "C" void kernel_launch(void* const* d_in, const int* in_sizes, int n_in,
                              void* d_out, int out_size, void* d_ws, size_t ws_size,
                              hipStream_t stream)
{
    const float* x  = (const float*)d_in[0];
    const float* wq = (const float*)d_in[1];
    const float* wk = (const float*)d_in[2];
    const float* wv = (const float*)d_in[3];

    _Float16* Qh = (_Float16*)d_ws;          // [SEQ][DOUT] f16 (pre-scaled)
    _Float16* Kh = Qh + (size_t)SEQ * DOUT;  // [SEQ][DOUT] f16
    _Float16* Vt = Kh + (size_t)SEQ * DOUT;  // [DOUT][SEQ] f16 (transposed)

    dim3 g1(SEQ / 64, 24);                   // 512 row tiles x (3 mats * 8 cols)
    qkv_kernel<<<g1, 128, 0, stream>>>(x, wq, wk, wv, Qh, Kh, Vt);

    attn_kernel<<<SEQ / 64, 128, 0, stream>>>(Qh, Kh, Vt, (float*)d_out);
}
